// RGIN_86114094285436
// MI455X (gfx1250) — matrix operand order repaired and checked by reading, not compiled, not run on hardware
//
#include <hip/hip_runtime.h>
#include <hip/hip_bf16.h>

// ---------------------------------------------------------------------------
// RGIN (3-layer relational GIN), N=40000, E=640000, D=H=128, R=8, B=4.
// Dense algebra via v_wmma_f32_16x16x32_bf16 (wave32 CDNA5); B-operand staged
// per block with global_load_async_to_lds_b128 (ASYNCcnt) and read via ds_load.
// ---------------------------------------------------------------------------

typedef __bf16 bf16;
typedef bf16  v16bf __attribute__((ext_vector_type(16)));
typedef bf16  v8bf  __attribute__((ext_vector_type(8)));
typedef bf16  v4bf  __attribute__((ext_vector_type(4)));
typedef float v8f   __attribute__((ext_vector_type(8)));

#define LN_EPS 1e-5f

// ------------------------------- utility kernels ---------------------------

__global__ __launch_bounds__(256) void zero_f32_kernel(float* __restrict__ p, int n) {
    int t = blockIdx.x * 256 + threadIdx.x;
    if (t < n) p[t] = 0.0f;
}

__global__ __launch_bounds__(256) void f32_to_bf16_kernel(const float* __restrict__ src,
                                                          bf16* __restrict__ dst, int n) {
    int t = blockIdx.x * 256 + threadIdx.x;
    if (t < n) dst[t] = (bf16)src[t];
}

// Per-(dst, relation) edge counts via native f32 atomics.
__global__ __launch_bounds__(256) void count_kernel(const int* __restrict__ edst,
                                                    const int* __restrict__ etype,
                                                    float* __restrict__ cnt, int E) {
    int e = blockIdx.x * 256 + threadIdx.x;
    if (e < E) unsafeAtomicAdd(&cnt[(size_t)edst[e] * 8 + etype[e]], 1.0f);
}

__global__ __launch_bounds__(256) void norm_kernel(const int* __restrict__ edst,
                                                   const int* __restrict__ etype,
                                                   const float* __restrict__ cnt,
                                                   float* __restrict__ nrm, int E) {
    int e = blockIdx.x * 256 + threadIdx.x;
    if (e < E) nrm[e] = 1.0f / fmaxf(cnt[(size_t)edst[e] * 8 + etype[e]], 1.0f);
}

// ------------------------- weight packing (B-fragment layout) ---------------
// Packed element t decomposes as: i = t&15 (half within lane), lane = (t>>4)&31,
// ks = (t>>9)&3 (K-step of 32), ct = t>>11 (16-col tile).
// B 32x16 layout used here (contiguous k order): col = ct*16 + (lane&15); K = ks*32 + (lane<16?0:16) + i.

__global__ __launch_bounds__(256) void pack_weight_kernel(const float* __restrict__ src,
                                                          bf16* __restrict__ dst,
                                                          int ncols, int total) {
    int t = blockIdx.x * 256 + threadIdx.x;
    if (t >= total) return;
    int i = t & 15, lane = (t >> 4) & 31, ks = (t >> 9) & 3, ct = t >> 11;
    int nn = ct * 16 + (lane & 15);
    int k  = ks * 32 + ((lane < 16) ? 0 : 16) + i;
    dst[t] = (bf16)src[(size_t)k * ncols + nn];
}

// Compose W[r] = sum_b comp[r,b] * wb[b] and pack [128, R*128] in one pass.
__global__ __launch_bounds__(256) void pack_composed_kernel(const float* __restrict__ comp,
                                                            const float* __restrict__ wb,
                                                            bf16* __restrict__ dst, int total) {
    int t = blockIdx.x * 256 + threadIdx.x;
    if (t >= total) return;
    int i = t & 15, lane = (t >> 4) & 31, ks = (t >> 9) & 3, ct = t >> 11;
    int nn = ct * 16 + (lane & 15);          // global col in [0,1024)
    int r  = nn >> 7, f = nn & 127;
    int k  = ks * 32 + ((lane < 16) ? 0 : 16) + i;
    float v = 0.0f;
#pragma unroll
    for (int b = 0; b < 4; ++b)
        v += comp[r * 4 + b] * wb[((size_t)b * 128 + k) * 128 + f];
    dst[t] = (bf16)v;
}

// ------------------------------ WMMA GEMM ----------------------------------
// C[M, Ncols] = A[M,128](bf16 row-major) x Bp(pre-packed fragments) + epilogue.
// Block: 256 threads = 8 waves; tile 128 rows x 64 cols; 16 WMMAs per wave.
// The block's B region is one contiguous 16KB slab of Bp -> staged once into
// LDS with async global->LDS b128 copies, consumed via ds_load_b128.

template <bool ADD_AGG, bool ADD_BIAS, bool RELU, bool STORE_F32, bool STORE_BF16>
__global__ __launch_bounds__(256) void gemm_bf16_wmma(
    const bf16* __restrict__ A, const bf16* __restrict__ Bp, int M,
    const float* __restrict__ agg, const float* __restrict__ xf32,
    const float* __restrict__ bias, const float* __restrict__ epsp,
    float* __restrict__ Df, bf16* __restrict__ Dbf, int ldd) {
    __shared__ bf16 bsh[8192];                 // 16 KB: 4 col-tiles x K=128

    const int lane = threadIdx.x & 31;
    const int wave = threadIdx.x >> 5;
    const int row0 = blockIdx.x * 128 + wave * 16;
    const int ct0  = blockIdx.y * 4;

    // ---- stage B fragments: contiguous 16KB -> LDS via async copy ----------
    {
        const bf16* gsrc = Bp + (size_t)ct0 * 2048;      // 2048 halves per col-tile
#pragma unroll
        for (int p = 0; p < 4; ++p) {
            const int c = p * 256 + threadIdx.x;         // 16-byte chunk id [0,1024)
            unsigned ldsaddr = (unsigned)(size_t)(&bsh[c * 8]);
            const bf16* gaddr = gsrc + c * 8;
            asm volatile("global_load_async_to_lds_b128 %0, %1, off"
                         :: "v"(ldsaddr), "v"(gaddr) : "memory");
        }
        asm volatile("s_wait_asynccnt 0x0" ::: "memory");
        __syncthreads();
    }

    // ---- preload the full A K-strip (K=128) for this wave's 16 rows --------
    // A 16x32 layout (matches packed B): row = lane&15; lanes 0-15 hold K 0..15,
    // lanes 16-31 hold K 16..31 per 32-K step.
    int m = row0 + (lane & 15);
    m = (m < M) ? m : (M - 1);
    const bf16* arow = A + (size_t)m * 128 + ((lane < 16) ? 0 : 16);
    v8bf afr[8];
#pragma unroll
    for (int ks = 0; ks < 4; ++ks) {
        afr[2 * ks]     = *(const v8bf*)(arow + ks * 32);
        afr[2 * ks + 1] = *(const v8bf*)(arow + ks * 32 + 8);
    }

    v8f acc[4] = {};
#pragma unroll
    for (int ks = 0; ks < 4; ++ks) {
        v16bf afrag = __builtin_shufflevector(afr[2 * ks], afr[2 * ks + 1],
                                              0, 1, 2, 3, 4, 5, 6, 7,
                                              8, 9, 10, 11, 12, 13, 14, 15);
#pragma unroll
        for (int t = 0; t < 4; ++t) {
            const bf16* bptr = &bsh[((t * 4 + ks) * 32 + lane) * 16];
            v8bf blo = *(const v8bf*)(bptr);
            v8bf bhi = *(const v8bf*)(bptr + 8);
            v16bf bfrag = __builtin_shufflevector(blo, bhi, 0, 1, 2, 3, 4, 5, 6, 7,
                                                  8, 9, 10, 11, 12, 13, 14, 15);
            acc[t] = __builtin_amdgcn_wmma_f32_16x16x32_bf16(
                false, afrag, false, bfrag, (short)0, acc[t], false, false);
        }
    }

    // ---- epilogue. C/D f32 layout: col = lane&15; row = j + (lane<16?0:8) ---
    const float eps1  = ADD_AGG ? (1.0f + epsp[0]) : 0.0f;
    const int   rbase = row0 + ((lane < 16) ? 0 : 8);
#pragma unroll
    for (int t = 0; t < 4; ++t) {
        const int col = (ct0 + t) * 16 + (lane & 15);
        const float bv = ADD_BIAS ? bias[col] : 0.0f;
#pragma unroll
        for (int j = 0; j < 8; ++j) {
            const int r = rbase + j;
            if (r < M) {
                float v = acc[t][j] + bv;
                size_t idx = (size_t)r * ldd + col;
                if (ADD_AGG) v += agg[idx] + eps1 * xf32[idx];
                if (RELU) v = fmaxf(v, 0.0f);
                if (STORE_F32) Df[idx] = v;
                if (STORE_BF16) Dbf[idx] = (bf16)v;
            }
        }
    }
}

// -------------------------- edge scatter (message agg) ----------------------
// One wave per edge; each lane handles 4 of 128 features; native f32 atomics.

__global__ __launch_bounds__(256) void scatter_kernel(const bf16* __restrict__ xr,
                                                      const int* __restrict__ esrc,
                                                      const int* __restrict__ etype,
                                                      const int* __restrict__ edst,
                                                      const float* __restrict__ nrm,
                                                      float* __restrict__ agg, int E) {
    int t = blockIdx.x * 256 + threadIdx.x;
    int e = t >> 5;
    if (e >= E) return;
    int lane = t & 31;
    int s = esrc[e], r = etype[e], d = edst[e];
    float w = nrm[e];
    v4bf msg = *(const v4bf*)(xr + ((size_t)s * 8 + r) * 128 + lane * 4);
    float* out = agg + (size_t)d * 128 + lane * 4;
#pragma unroll
    for (int i = 0; i < 4; ++i)
        unsafeAtomicAdd(&out[i], w * (float)msg[i]);
}

// ----------------------- LayerNorm + ReLU (wave per row) --------------------

__global__ __launch_bounds__(256) void ln_relu_kernel(const float* __restrict__ h,
                                                      const float* __restrict__ g,
                                                      const float* __restrict__ be,
                                                      bf16* __restrict__ out, int M) {
    int wave = threadIdx.x >> 5, lane = threadIdx.x & 31;
    int row = blockIdx.x * 8 + wave;
    if (row >= M) return;
    const float4 v = *(const float4*)(h + (size_t)row * 128 + lane * 4);
    float s = v.x + v.y + v.z + v.w;
    float q = v.x * v.x + v.y * v.y + v.z * v.z + v.w * v.w;
#pragma unroll
    for (int off = 16; off >= 1; off >>= 1) {
        s += __shfl_xor(s, off, 32);
        q += __shfl_xor(q, off, 32);
    }
    const float mean = s * (1.0f / 128.0f);
    const float var  = q * (1.0f / 128.0f) - mean * mean;
    const float rs   = rsqrtf(var + LN_EPS);
    const int col = lane * 4;
    v4bf o;
    o[0] = (bf16)fmaxf((v.x - mean) * rs * g[col + 0] + be[col + 0], 0.0f);
    o[1] = (bf16)fmaxf((v.y - mean) * rs * g[col + 1] + be[col + 1], 0.0f);
    o[2] = (bf16)fmaxf((v.z - mean) * rs * g[col + 2] + be[col + 2], 0.0f);
    o[3] = (bf16)fmaxf((v.w - mean) * rs * g[col + 3] + be[col + 3], 0.0f);
    *(v4bf*)(out + (size_t)row * 128 + col) = o;
}

// --------------------------------- host side -------------------------------

extern "C" void kernel_launch(void* const* d_in, const int* in_sizes, int n_in,
                              void* d_out, int out_size, void* d_ws, size_t ws_size,
                              hipStream_t stream) {
    const int D = 128, R = 8;
    const int N = in_sizes[0] / D;
    const int E = in_sizes[2];

    const float* x     = (const float*)d_in[0];
    const int*   eidx  = (const int*)d_in[1];
    const int*   etype = (const int*)d_in[2];
    const int*   esrc  = eidx;
    const int*   edst  = eidx + E;

    // params per layer: 0 wb, 1 comp, 2 root, 3 bias, 4 eps, 5 w1, 6 b1, 7 g, 8 be, 9 w2, 10 b2
    auto P = [&](int layer, int which) -> const float* {
        return (const float*)d_in[3 + layer * 11 + which];
    };

    char* wsp = (char*)d_ws;
    auto alloc = [&](size_t bytes) -> char* {
        char* p = wsp;
        wsp += (bytes + 255) & ~(size_t)255;
        return p;
    };
    bf16*  xbf_in = (bf16*)alloc((size_t)N * D * 2);
    bf16*  ybf_a  = (bf16*)alloc((size_t)N * D * 2);
    bf16*  ybf_b  = (bf16*)alloc((size_t)N * D * 2);
    float* yf_a   = (float*)alloc((size_t)N * D * 4);
    float* yf_b   = (float*)alloc((size_t)N * D * 4);
    bf16*  xr     = (bf16*)alloc((size_t)N * R * D * 2);   // [N, R, 128] bf16
    float* agg    = (float*)alloc((size_t)N * D * 4);
    bf16*  out_bf = (bf16*)alloc((size_t)N * D * 2);
    float* hbuf   = (float*)alloc((size_t)N * D * 4);
    bf16*  hbf    = (bf16*)alloc((size_t)N * D * 2);
    float* cnt    = (float*)alloc((size_t)N * R * 4);
    float* nrm    = (float*)alloc((size_t)E * 4);
    bf16*  Wcatp  = (bf16*)alloc((size_t)R * D * D * 2);
    bf16*  rootp  = (bf16*)alloc((size_t)D * D * 2);
    bf16*  w1p    = (bf16*)alloc((size_t)D * D * 2);
    bf16*  w2p    = (bf16*)alloc((size_t)D * D * 2);

    const int mblocks = (N + 127) / 128;
    const int pk_big = R * D * D;   // 131072 packed elements
    const int pk_sm  = D * D;       // 16384

    // One-time setup: degree normalization + bf16 input activations.
    zero_f32_kernel<<<(N * R + 255) / 256, 256, 0, stream>>>(cnt, N * R);
    count_kernel<<<(E + 255) / 256, 256, 0, stream>>>(edst, etype, cnt, E);
    norm_kernel<<<(E + 255) / 256, 256, 0, stream>>>(edst, etype, cnt, nrm, E);
    f32_to_bf16_kernel<<<(N * D + 255) / 256, 256, 0, stream>>>(x, xbf_in, N * D);

    const float* xf = x;
    const bf16*  xb = xbf_in;

    for (int l = 0; l < 3; ++l) {
        // Pack composed relation weights + dense weights into fragment layout.
        pack_composed_kernel<<<(pk_big + 255) / 256, 256, 0, stream>>>(P(l, 1), P(l, 0), Wcatp, pk_big);
        pack_weight_kernel<<<(pk_sm + 255) / 256, 256, 0, stream>>>(P(l, 2), rootp, D, pk_sm);
        pack_weight_kernel<<<(pk_sm + 255) / 256, 256, 0, stream>>>(P(l, 5), w1p, D, pk_sm);
        pack_weight_kernel<<<(pk_sm + 255) / 256, 256, 0, stream>>>(P(l, 9), w2p, D, pk_sm);
        zero_f32_kernel<<<(N * D + 255) / 256, 256, 0, stream>>>(agg, N * D);

        // xr[n, r*128+f] = sum_d x[n,d] W[r,d,f]   (dominant GEMM, [N,128]x[128,1024])
        gemm_bf16_wmma<false, false, false, false, true>
            <<<dim3(mblocks, 16), 256, 0, stream>>>(xb, Wcatp, N, nullptr, nullptr,
                                                    nullptr, nullptr, nullptr, xr, R * D);

        // agg[dst] += norm * xr[src, etype]
        scatter_kernel<<<(int)(((size_t)E * 32 + 255) / 256), 256, 0, stream>>>(
            xr, esrc, etype, edst, nrm, agg, E);

        // out = agg + x@root + bias + (1+eps)*x   -> bf16 for MLP input
        gemm_bf16_wmma<true, true, false, false, true>
            <<<dim3(mblocks, 2), 256, 0, stream>>>(xb, rootp, N, agg, xf,
                                                   P(l, 3), P(l, 4), nullptr, out_bf, D);

        // h = out@w1 + b1 (f32, pre-LayerNorm)
        gemm_bf16_wmma<false, true, false, true, false>
            <<<dim3(mblocks, 2), 256, 0, stream>>>(out_bf, w1p, N, nullptr, nullptr,
                                                   P(l, 6), nullptr, hbuf, nullptr, D);

        // LayerNorm + ReLU -> bf16
        ln_relu_kernel<<<(N + 7) / 8, 256, 0, stream>>>(hbuf, P(l, 7), P(l, 8), hbf, N);

        // y = h@w2 + b2 (+ReLU between layers)
        if (l < 2) {
            float* yo  = (l == 0) ? yf_a : yf_b;
            bf16*  yob = (l == 0) ? ybf_a : ybf_b;
            gemm_bf16_wmma<false, true, true, true, true>
                <<<dim3(mblocks, 2), 256, 0, stream>>>(hbf, w2p, N, nullptr, nullptr,
                                                       P(l, 10), nullptr, yo, yob, D);
            xf = yo;
            xb = yob;
        } else {
            gemm_bf16_wmma<false, true, false, true, false>
                <<<dim3(mblocks, 2), 256, 0, stream>>>(hbf, w2p, N, nullptr, nullptr,
                                                       P(l, 10), nullptr, (float*)d_out,
                                                       nullptr, D);
        }
    }
}
